// MultiHead_3341484556940
// MI455X (gfx1250) — compile-verified, matrix operand order checked by reading, not run on hardware
//
#include <hip/hip_runtime.h>
#include <hip/hip_bf16.h>

typedef __attribute__((ext_vector_type(16))) __bf16 v16bf;
typedef __attribute__((ext_vector_type(8)))  float  v8f;

#define B_NUM 4
#define T_SEQ 1024
#define C_DIM 768
#define H_NUM 12
#define D_DIM 64
#define TT_NUM (T_SEQ / 16)        // 64 query/key 16-row tiles
#define KSTEPS (C_DIM / 32)        // 24 K-steps for projections

// element counts of swizzled workspace regions (bf16 elements)
#define XS_ELEMS (B_NUM * TT_NUM * KSTEPS * 512)          // 3,145,728
#define WS_ELEMS (3 * H_NUM * 4 * KSTEPS * 512)           // 1,769,472
#define QS_ELEMS (B_NUM * H_NUM * TT_NUM * 2 * 512)       // 3,145,728
#define VS_ELEMS (B_NUM * H_NUM * 32 * 4 * 512)           // 3,145,728

// K index held by (lane, e) in a 16-bit A/B WMMA fragment (16x32 / 32x16)
__device__ __forceinline__ int kmap(int lane, int e) {
    int p8 = (lane & 16) ? 8 : 0;
    return (e < 8) ? (p8 + e) : (16 + p8 + (e - 8));
}

__device__ __forceinline__ v8f wmma_bf16(v16bf a, v16bf b, v8f c) {
    return __builtin_amdgcn_wmma_f32_16x16x32_bf16(false, a, false, b, (short)0, c, false, false);
}

// ---------------------------------------------------------------------------
// Kernel 0a: x[B,T,C] f32 -> A-operand bf16 fragments [B*TT, KSTEPS, lane, e]
// ---------------------------------------------------------------------------
__global__ void swz_x(const float* __restrict__ x, __bf16* __restrict__ xs, int total) {
    int idx = blockIdx.x * blockDim.x + threadIdx.x;
    if (idx >= total) return;
    int e     = idx & 15;
    int lane  = (idx >> 4) & 31;
    int kstep = (idx >> 9) % KSTEPS;
    int bt    = idx / (KSTEPS * 512);          // b*TT_NUM + tt
    int b  = bt >> 6;
    int tt = bt & 63;
    int t = tt * 16 + (lane & 15);
    int c = kstep * 32 + kmap(lane, e);
    xs[idx] = (__bf16)x[((size_t)b * T_SEQ + t) * C_DIM + c];
}

// ---------------------------------------------------------------------------
// Kernel 0b: Wq/Wk/Wv [H,C,D] f32 -> B-operand bf16 fragments
//            layout [m(3), h, ntile(4), kstep(24), lane, e]
// ---------------------------------------------------------------------------
__global__ void swz_w(const float* __restrict__ Wq, const float* __restrict__ Wk,
                      const float* __restrict__ Wv, __bf16* __restrict__ wsz, int total) {
    int idx = blockIdx.x * blockDim.x + threadIdx.x;
    if (idx >= total) return;
    int e     = idx & 15;
    int lane  = (idx >> 4) & 31;
    int kstep = (idx >> 9) % KSTEPS;
    int rest  = idx / (KSTEPS * 512);          // m*H*4 + h*4 + ntile
    int ntile = rest & 3;
    int h     = (rest >> 2) % H_NUM;
    int m     = rest / (4 * H_NUM);
    const float* W = (m == 0) ? Wq : ((m == 1) ? Wk : Wv);
    int k = kstep * 32 + kmap(lane, e);
    int n = ntile * 16 + (lane & 15);
    wsz[idx] = (__bf16)W[((size_t)h * C_DIM + k) * D_DIM + n];
}

// ---------------------------------------------------------------------------
// Kernel 1: QKV projection.  grid=(B*TT, H), block=384 (12 waves).
// wave w: m = w>>2 selects q/k/v, ntile = w&3 selects 16 of 64 output cols.
// Results staged through LDS and written back pre-swizzled for the attention
// kernel: Q,K as (B-op for Q^T / A-op for K) fragments, V as V^T A-op frags.
// ---------------------------------------------------------------------------
__global__ __launch_bounds__(384) void qkv_proj(
    const __bf16* __restrict__ xs, const __bf16* __restrict__ wsz,
    __bf16* __restrict__ q_sw, __bf16* __restrict__ k_sw, __bf16* __restrict__ v_sw) {

    __shared__ float tile[3][16][D_DIM + 4];

    int bt   = blockIdx.x;                 // b*TT + tt
    int h    = blockIdx.y;
    int tid  = threadIdx.x;
    int w    = tid >> 5;
    int lane = tid & 31;
    int m = w >> 2, ntile = w & 3;

    const __bf16* aBase = xs  + (size_t)bt * (KSTEPS * 512) + lane * 16;
    const __bf16* bBase = wsz + (size_t)((m * H_NUM + h) * 4 + ntile) * (KSTEPS * 512) + lane * 16;

    v8f acc = {0.f, 0.f, 0.f, 0.f, 0.f, 0.f, 0.f, 0.f};
    for (int s = 0; s < KSTEPS; ++s) {
        v16bf a = *(const v16bf*)(aBase + s * 512);
        v16bf b = *(const v16bf*)(bBase + s * 512);
        if (s + 1 < KSTEPS) __builtin_prefetch(bBase + (s + 1) * 512, 0, 1);
        acc = wmma_bf16(a, b, acc);
    }

    // D fragment -> LDS plain tile (lane holds col n, rows r+8*half)
    int p8 = (lane & 16) ? 8 : 0;
    int n  = ntile * 16 + (lane & 15);
    #pragma unroll
    for (int r = 0; r < 8; ++r) tile[m][r + p8][n] = acc[r];
    __syncthreads();

    int bh = (bt >> 6) * H_NUM + h;        // b*H + h
    int tt = bt & 63;

    for (int j = tid; j < 3 * 1024; j += 384) {
        int sel = j >> 10;
        int j2  = j & 1023;
        if (sel < 2) {
            // Q / K: [bh, tt, dstep(2), lane, e] ; value = M[t0+(l&15)][dmap]
            int e = j2 & 15, l = (j2 >> 4) & 31, dstep = j2 >> 9;
            float v = tile[sel][l & 15][dstep * 32 + kmap(l, e)];
            __bf16* dst = (sel == 0) ? q_sw : k_sw;
            dst[((size_t)bh * TT_NUM + tt) * 1024 + j2] = (__bf16)v;
        } else {
            // V^T A-op frags: [bh, sstep(32), dtile(4), lane, e]
            int dt = j2 >> 8, rem = j2 & 255, l = rem >> 3, ep = rem & 7;
            int lp8 = (l & 16) ? 8 : 0;
            float v = tile[2][lp8 + ep][dt * 16 + (l & 15)];
            int ss = tt >> 1, eh = tt & 1;
            v_sw[(((size_t)bh * 32 + ss) * 4 + dt) * 512 + l * 16 + (ep + 8 * eh)] = (__bf16)v;
        }
    }
}

// ---------------------------------------------------------------------------
// Kernel 2: flash attention, one wave per (b,h,16-query tile).
// Computes S^T = K*Q^T so exp(S) repacks in-lane straight into the B-operand
// of out^T = V^T * P^T (no cross-lane transpose between the two GEMMs).
// ---------------------------------------------------------------------------
__global__ __launch_bounds__(32) void attn(
    const __bf16* __restrict__ q_sw, const __bf16* __restrict__ k_sw,
    const __bf16* __restrict__ v_sw, float* __restrict__ out) {

    int blk = blockIdx.x;
    int tt  = blk & 63;
    int bh  = blk >> 6;
    int b = bh / H_NUM, h = bh % H_NUM;
    int lane = threadIdx.x;
    int p8   = (lane & 16) ? 8 : 0;
    int tcol = tt * 16 + (lane & 15);      // this lane's query index
    const float scale = 0.125f;            // 1/sqrt(64)
    const float NEG = -__builtin_inff();

    const __bf16* qb_p = q_sw + ((size_t)bh * TT_NUM + tt) * 1024 + lane * 16;
    v16bf qb0 = *(const v16bf*)(qb_p);
    v16bf qb1 = *(const v16bf*)(qb_p + 512);

    float m_run = NEG, l_run = 0.f;
    v8f o0 = {0.f,0.f,0.f,0.f,0.f,0.f,0.f,0.f};
    v8f o1 = o0, o2 = o0, o3 = o0;

    int nchunks = (tt >> 1) + 1;           // 32 keys per chunk, causal bound
    for (int c = 0; c < nchunks; ++c) {
        int s0 = c * 32;
        const __bf16* kb = k_sw + ((size_t)bh * TT_NUM + 2 * c) * 1024 + lane * 16;
        v16bf ka00 = *(const v16bf*)(kb);
        v16bf ka01 = *(const v16bf*)(kb + 512);
        v16bf ka10 = *(const v16bf*)(kb + 1024);
        v16bf ka11 = *(const v16bf*)(kb + 1536);

        v8f sc0 = {0.f,0.f,0.f,0.f,0.f,0.f,0.f,0.f};
        v8f sc1 = sc0;
        sc0 = wmma_bf16(ka00, qb0, sc0);   // S^T rows s0..s0+15
        sc0 = wmma_bf16(ka01, qb1, sc0);
        sc1 = wmma_bf16(ka10, qb0, sc1);   // S^T rows s0+16..s0+31
        sc1 = wmma_bf16(ka11, qb1, sc1);

        // scale + causal mask (row index = key s, col = query t)
        float x0[8], x1[8];
        float cmax = NEG;
        #pragma unroll
        for (int r = 0; r < 8; ++r) {
            int srow = s0 + p8 + r;
            x0[r] = (srow      <= tcol) ? sc0[r] * scale : NEG;
            x1[r] = (srow + 16 <= tcol) ? sc1[r] * scale : NEG;
            cmax = fmaxf(cmax, fmaxf(x0[r], x1[r]));
        }
        cmax = fmaxf(cmax, __shfl_xor(cmax, 16, 32));   // other 16 key rows of this column
        float new_m = fmaxf(m_run, cmax);
        float corr  = __expf(m_run - new_m);

        float csum = 0.f;
        v16bf pb;                                        // P^T B-operand, in-lane repack
        #pragma unroll
        for (int r = 0; r < 8; ++r) {
            float e0 = __expf(x0[r] - new_m);
            float e1 = __expf(x1[r] - new_m);
            csum += e0 + e1;
            pb[r]     = (__bf16)e0;
            pb[r + 8] = (__bf16)e1;
        }
        csum += __shfl_xor(csum, 16, 32);
        l_run = l_run * corr + csum;
        m_run = new_m;

        #pragma unroll
        for (int r = 0; r < 8; ++r) {
            o0[r] *= corr; o1[r] *= corr; o2[r] *= corr; o3[r] *= corr;
        }

        const __bf16* vb = v_sw + (((size_t)bh * 32 + c) * 4) * 512 + lane * 16;
        o0 = wmma_bf16(*(const v16bf*)(vb),        pb, o0);
        o1 = wmma_bf16(*(const v16bf*)(vb + 512),  pb, o1);
        o2 = wmma_bf16(*(const v16bf*)(vb + 1024), pb, o2);
        o3 = wmma_bf16(*(const v16bf*)(vb + 1536), pb, o3);
    }

    float inv = 1.0f / l_run;
    // out[b][t][h*64 + d];  lane holds col t = tcol, rows d = dt*16 + p8 + r
    float* op = out + ((size_t)b * T_SEQ + tcol) * (H_NUM * D_DIM) + h * D_DIM;
    #pragma unroll
    for (int r = 0; r < 8; ++r) {
        op[      p8 + r] = o0[r] * inv;
        op[16 +  p8 + r] = o1[r] * inv;
        op[32 +  p8 + r] = o2[r] * inv;
        op[48 +  p8 + r] = o3[r] * inv;
    }
}

// ---------------------------------------------------------------------------
extern "C" void kernel_launch(void* const* d_in, const int* in_sizes, int n_in,
                              void* d_out, int out_size, void* d_ws, size_t ws_size,
                              hipStream_t stream) {
    const float* x  = (const float*)d_in[0];
    const float* Wq = (const float*)d_in[1];
    const float* Wk = (const float*)d_in[2];
    const float* Wv = (const float*)d_in[3];
    float* out = (float*)d_out;

    char* ws = (char*)d_ws;
    __bf16* xs   = (__bf16*)(ws);
    __bf16* wsz  = (__bf16*)(ws + (size_t)XS_ELEMS * 2);
    __bf16* q_sw = (__bf16*)(ws + ((size_t)XS_ELEMS + WS_ELEMS) * 2);
    __bf16* k_sw = (__bf16*)(ws + ((size_t)XS_ELEMS + WS_ELEMS + QS_ELEMS) * 2);
    __bf16* v_sw = (__bf16*)(ws + ((size_t)XS_ELEMS + WS_ELEMS + 2 * (size_t)QS_ELEMS) * 2);

    swz_x<<<(XS_ELEMS + 255) / 256, 256, 0, stream>>>(x, xs, XS_ELEMS);
    swz_w<<<(WS_ELEMS + 255) / 256, 256, 0, stream>>>(Wq, Wk, Wv, wsz, WS_ELEMS);

    dim3 g1(B_NUM * TT_NUM, H_NUM);
    qkv_proj<<<g1, 384, 0, stream>>>(xs, wsz, q_sw, k_sw, v_sw);

    attn<<<B_NUM * H_NUM * TT_NUM, 32, 0, stream>>>(q_sw, k_sw, v_sw, out);
}
